// SelfAttention_6236292514086
// MI455X (gfx1250) — compile-verified, matrix operand order checked by reading, not run on hardware
//
#include <hip/hip_runtime.h>

#define B_  8
#define T_  1024
#define E_  256
#define H_  8
#define EH_ (E_ * H_)   // 2048
#define BT_ (B_ * T_)   // 8192

typedef __bf16 bf16_t;
typedef bf16_t bf16x16 __attribute__((ext_vector_type(16)));
typedef float  f32x8   __attribute__((ext_vector_type(8)));
typedef unsigned int u32x4 __attribute__((ext_vector_type(4)));

union FragU { bf16x16 v; u32x4 q[2]; };

static_assert(sizeof(bf16x16) == 32, "bf16x16 must be 32 bytes");

__device__ __forceinline__ unsigned short f2bf(float f) {
    unsigned int u = __float_as_uint(f);
    u += 0x7FFFu + ((u >> 16) & 1u);          // round-to-nearest-even
    return (unsigned short)(u >> 16);
}

// Load a 16x32 bf16 operand fragment (A- or B-style: operand stored as
// rows x K, K contiguous).  Lane l: row = row0 + (l&15); K chunks at
// k0 + (l>>4)*8 and k0 + 16 + (l>>4)*8  (two global_load_b128).
__device__ __forceinline__ bf16x16 ld_frag_g(const unsigned short* __restrict__ base,
                                             int ld, int row0, int k0, int lane) {
    const unsigned short* p = base + (row0 + (lane & 15)) * ld + k0 + ((lane >> 4) << 3);
    FragU u;
    u.q[0] = *reinterpret_cast<const u32x4*>(p);
    u.q[1] = *reinterpret_cast<const u32x4*>(p + 16);
    return u.v;
}

// Same pattern but from LDS (row0 == 0; tile base passed directly).
__device__ __forceinline__ bf16x16 ld_frag_s(const unsigned short* base,
                                             int ld, int k0, int lane) {
    const unsigned short* p = base + (lane & 15) * ld + k0 + ((lane >> 4) << 3);
    FragU u;
    u.q[0] = *reinterpret_cast<const u32x4*>(p);
    u.q[1] = *reinterpret_cast<const u32x4*>(p + 16);
    return u.v;
}

__device__ __forceinline__ f32x8 wmma_bf16(bf16x16 a, bf16x16 b, f32x8 c) {
    return __builtin_amdgcn_wmma_f32_16x16x32_bf16(false, a, false, b, (short)0, c,
                                                   false, false);
}

// ---------------- pack kernels ----------------

__global__ void k_cvt(const float* __restrict__ src, unsigned short* __restrict__ dst, int n) {
    int i = blockIdx.x * 256 + threadIdx.x;
    if (i < n) dst[i] = f2bf(src[i]);
}

// dst[n*K + k] = bf16(src[k*N + n] * scale)   (transpose to N x K, K contiguous)
__global__ void k_tpack(const float* __restrict__ src, unsigned short* __restrict__ dst,
                        int K, int N, float scale) {
    int i = blockIdx.x * 256 + threadIdx.x;
    if (i >= K * N) return;
    int n = i / K;
    int k = i - n * K;
    dst[i] = f2bf(src[k * N + n] * scale);
}

// ---------------- QKV projection ----------------
// C(BT x EH) = Xbf(BT x E) @ Wt(EH x E)^T ; blockIdx.z selects Q/K/V.
// Q,K stored (b,h,t,e);  V stored transposed (b,h,e,t).
__global__ __launch_bounds__(128) void k_qkv(const unsigned short* __restrict__ Xbf,
                      const unsigned short* __restrict__ Wqt,
                      const unsigned short* __restrict__ Wkt,
                      const unsigned short* __restrict__ Wvt,
                      unsigned short* __restrict__ Qbf,
                      unsigned short* __restrict__ Kbf,
                      unsigned short* __restrict__ Vtb) {
    int lane = threadIdx.x;
    int mt = blockIdx.x;                         // 16-row tile over BT
    int nt = blockIdx.y * 4 + threadIdx.y;       // 16-col tile over EH
    int z  = blockIdx.z;
    const unsigned short* Wt = (z == 0) ? Wqt : (z == 1) ? Wkt : Wvt;

    f32x8 acc = {0.f, 0.f, 0.f, 0.f, 0.f, 0.f, 0.f, 0.f};
#pragma unroll
    for (int kk = 0; kk < 8; ++kk) {
        bf16x16 a = ld_frag_g(Xbf, E_, mt * 16, kk * 32, lane);
        bf16x16 b = ld_frag_g(Wt,  E_, nt * 16, kk * 32, lane);
        acc = wmma_bf16(a, b, acc);
    }

    int hf = lane >> 4, nl = lane & 15;
    int n  = nt * 16 + nl;
    int h  = n >> 8;                             // n / 256
    int e  = n & 255;
#pragma unroll
    for (int v = 0; v < 8; ++v) {
        int m  = mt * 16 + v + hf * 8;
        int bb = m >> 10;                        // m / 1024
        int tt = m & 1023;
        unsigned short val = f2bf(acc[v]);
        if (z == 0)      Qbf[((bb * H_ + h) * T_ + tt) * E_ + e] = val;
        else if (z == 1) Kbf[((bb * H_ + h) * T_ + tt) * E_ + e] = val;
        else             Vtb[((bb * H_ + h) * E_ + e) * T_ + tt] = val;
    }
}

// ---------------- fused attention ----------------
// One workgroup (8 waves) per (b, h, 16-query-row tile).
// Phase 1: wave w computes scores for key columns [w*128, w*128+128).
// Phase 2: cross-wave masked softmax via shuffles + LDS.
// Phase 3: attn tile (bf16) in LDS; O = attn @ V  (wave w -> e in [w*32, w*32+32)).
__global__ __launch_bounds__(256) void k_attn(const unsigned short* __restrict__ Qbf,
                       const unsigned short* __restrict__ Kbf,
                       const unsigned short* __restrict__ Vtb,
                       const int* __restrict__ mask,
                       unsigned short* __restrict__ Obf) {
    __shared__ unsigned short attn_s[16 * 1024];   // 32 KB probs tile
    __shared__ float red_s[16 * 8];                // per-row per-wave partials
    __shared__ float stat_s[16];                   // per-row max / inv-sum

    int lane = threadIdx.x & 31;
    int w    = threadIdx.x >> 5;
    int qt = blockIdx.x, hh = blockIdx.y, bb = blockIdx.z;
    int hf = lane >> 4, nl = lane & 15;

    const unsigned short* Qh = Qbf + (size_t)((bb * H_ + hh) * T_) * E_;
    const unsigned short* Kh = Kbf + (size_t)((bb * H_ + hh) * T_) * E_;
    const unsigned short* Vh = Vtb + (size_t)((bb * H_ + hh) * E_) * T_;
    const int* mb = mask + bb * T_;

    int mi[8], mj[8];
#pragma unroll
    for (int v = 0; v < 8; ++v)  mi[v]  = mb[qt * 16 + v + hf * 8];
#pragma unroll
    for (int nt = 0; nt < 8; ++nt) mj[nt] = mb[w * 128 + nt * 16 + nl];

    // Q fragments for this 16-row tile (K = 256 -> 8 chained wmma steps)
    bf16x16 qf[8];
#pragma unroll
    for (int kk = 0; kk < 8; ++kk) qf[kk] = ld_frag_g(Qh, E_, qt * 16, kk * 32, lane);

    // scores: 8 tiles of 16x16 per wave
    f32x8 acc[8];
#pragma unroll
    for (int nt = 0; nt < 8; ++nt) {
        f32x8 c = {0.f, 0.f, 0.f, 0.f, 0.f, 0.f, 0.f, 0.f};
#pragma unroll
        for (int kk = 0; kk < 8; ++kk) {
            bf16x16 kfr = ld_frag_g(Kh, E_, w * 128 + nt * 16, kk * 32, lane);
            c = wmma_bf16(qf[kk], kfr, c);
        }
        acc[nt] = c;
    }

    // per-row max (masked)
    float pmax[8];
#pragma unroll
    for (int v = 0; v < 8; ++v) pmax[v] = -1e30f;
#pragma unroll
    for (int nt = 0; nt < 8; ++nt)
#pragma unroll
        for (int v = 0; v < 8; ++v) {
            float lg = (mi[v] & mj[nt]) ? acc[nt][v] : -1e30f;
            pmax[v] = fmaxf(pmax[v], lg);
        }
#pragma unroll
    for (int off = 1; off < 16; off <<= 1)
#pragma unroll
        for (int v = 0; v < 8; ++v)
            pmax[v] = fmaxf(pmax[v], __shfl_xor(pmax[v], off, 32));
    if (nl == 0) {
#pragma unroll
        for (int v = 0; v < 8; ++v) red_s[(v + hf * 8) * 8 + w] = pmax[v];
    }
    __syncthreads();
    if (threadIdx.x < 16) {
        float m = -1e30f;
        for (int ww = 0; ww < 8; ++ww) m = fmaxf(m, red_s[threadIdx.x * 8 + ww]);
        stat_s[threadIdx.x] = m;
    }
    __syncthreads();

    float rmax[8];
#pragma unroll
    for (int v = 0; v < 8; ++v) rmax[v] = stat_s[v + hf * 8];

    // exp + per-row sum (overwrite acc with exp values)
    float psum[8] = {0.f, 0.f, 0.f, 0.f, 0.f, 0.f, 0.f, 0.f};
#pragma unroll
    for (int nt = 0; nt < 8; ++nt)
#pragma unroll
        for (int v = 0; v < 8; ++v) {
            float lg = (mi[v] & mj[nt]) ? acc[nt][v] : -1e30f;
            float ex = __expf(lg - rmax[v]);
            acc[nt][v] = ex;
            psum[v] += ex;
        }
#pragma unroll
    for (int off = 1; off < 16; off <<= 1)
#pragma unroll
        for (int v = 0; v < 8; ++v) psum[v] += __shfl_xor(psum[v], off, 32);
    if (nl == 0) {
#pragma unroll
        for (int v = 0; v < 8; ++v) red_s[(v + hf * 8) * 8 + w] = psum[v];
    }
    __syncthreads();
    if (threadIdx.x < 16) {
        float s = 0.f;
        for (int ww = 0; ww < 8; ++ww) s += red_s[threadIdx.x * 8 + ww];
        stat_s[threadIdx.x] = 1.0f / s;
    }
    __syncthreads();

    float rinv[8];
#pragma unroll
    for (int v = 0; v < 8; ++v) rinv[v] = stat_s[v + hf * 8];

    // normalized probs -> LDS (bf16, row-major 16 x 1024)
#pragma unroll
    for (int nt = 0; nt < 8; ++nt)
#pragma unroll
        for (int v = 0; v < 8; ++v)
            attn_s[(v + hf * 8) * 1024 + w * 128 + nt * 16 + nl] =
                f2bf(acc[nt][v] * rinv[v]);
    __syncthreads();

    // O = attn @ V : K = 1024 -> 32 wmma steps; wave w covers 32 e-columns
    f32x8 oacc[2];
    oacc[0] = f32x8{0.f, 0.f, 0.f, 0.f, 0.f, 0.f, 0.f, 0.f};
    oacc[1] = f32x8{0.f, 0.f, 0.f, 0.f, 0.f, 0.f, 0.f, 0.f};
#pragma unroll 4
    for (int kk = 0; kk < 32; ++kk) {
        bf16x16 af = ld_frag_s(attn_s, 1024, kk * 32, lane);
#pragma unroll
        for (int j = 0; j < 2; ++j) {
            bf16x16 vf = ld_frag_g(Vh, T_, w * 32 + j * 16, kk * 32, lane);
            oacc[j] = wmma_bf16(af, vf, oacc[j]);
        }
    }

    // store O in (b, t, h, e) layout (K contiguous for the final GEMM)
#pragma unroll
    for (int j = 0; j < 2; ++j)
#pragma unroll
        for (int v = 0; v < 8; ++v) {
            int tt = qt * 16 + v + hf * 8;
            int ee = w * 32 + j * 16 + nl;
            Obf[((bb * T_ + tt) * H_ + hh) * E_ + ee] = f2bf(oacc[j][v]);
        }
}

// ---------------- output projection ----------------
// out(BT x E) f32 = Obf(BT x 2048) @ Wut(E x 2048)^T + bu
__global__ __launch_bounds__(128) void k_final(const unsigned short* __restrict__ Obf,
                        const unsigned short* __restrict__ Wut,
                        const float* __restrict__ bu,
                        float* __restrict__ out) {
    int lane = threadIdx.x;
    int mt = blockIdx.x;
    int nt = blockIdx.y * 4 + threadIdx.y;

    f32x8 acc = {0.f, 0.f, 0.f, 0.f, 0.f, 0.f, 0.f, 0.f};
#pragma unroll 8
    for (int kk = 0; kk < 64; ++kk) {
        bf16x16 a = ld_frag_g(Obf, EH_, mt * 16, kk * 32, lane);
        bf16x16 b = ld_frag_g(Wut, EH_, nt * 16, kk * 32, lane);
        acc = wmma_bf16(a, b, acc);
    }

    int hf = lane >> 4, nl = lane & 15;
    int n = nt * 16 + nl;
    float bias = bu[n];
#pragma unroll
    for (int v = 0; v < 8; ++v) {
        int m = mt * 16 + v + hf * 8;
        out[m * E_ + n] = acc[v] + bias;
    }
}

// ---------------- launcher ----------------
extern "C" void kernel_launch(void* const* d_in, const int* in_sizes, int n_in,
                              void* d_out, int out_size, void* d_ws, size_t ws_size,
                              hipStream_t stream) {
    (void)in_sizes; (void)n_in; (void)out_size; (void)ws_size;
    const float* x    = (const float*)d_in[0];
    const int*   mask = (const int*)d_in[1];
    const float* Wq   = (const float*)d_in[2];
    const float* Wk   = (const float*)d_in[3];
    const float* Wv   = (const float*)d_in[4];
    const float* Wu   = (const float*)d_in[5];
    const float* bu   = (const float*)d_in[6];
    float* out = (float*)d_out;

    char* ws = (char*)d_ws;
    // bf16 workspace layout (total 136 MB)
    unsigned short* Xbf = (unsigned short*)(ws);                        //  4 MB
    unsigned short* Wqt = (unsigned short*)(ws + (4ull  << 20));        //  1 MB
    unsigned short* Wkt = (unsigned short*)(ws + (5ull  << 20));        //  1 MB
    unsigned short* Wvt = (unsigned short*)(ws + (6ull  << 20));        //  1 MB
    unsigned short* Wut = (unsigned short*)(ws + (7ull  << 20));        //  1 MB
    unsigned short* Qbf = (unsigned short*)(ws + (8ull  << 20));        // 32 MB
    unsigned short* Kbf = (unsigned short*)(ws + (40ull << 20));        // 32 MB
    unsigned short* Vtb = (unsigned short*)(ws + (72ull << 20));        // 32 MB
    unsigned short* Obf = (unsigned short*)(ws + (104ull << 20));       // 32 MB

    // pack inputs to bf16 (Wq/Wk pre-scaled by 1/e^0.25 = 1/4 each)
    k_cvt  <<<(BT_ * E_) / 256, 256, 0, stream>>>(x, Xbf, BT_ * E_);
    k_tpack<<<(EH_ * E_) / 256, 256, 0, stream>>>(Wq, Wqt, E_,  EH_, 0.25f);
    k_tpack<<<(EH_ * E_) / 256, 256, 0, stream>>>(Wk, Wkt, E_,  EH_, 0.25f);
    k_tpack<<<(EH_ * E_) / 256, 256, 0, stream>>>(Wv, Wvt, E_,  EH_, 1.0f);
    k_tpack<<<(EH_ * E_) / 256, 256, 0, stream>>>(Wu, Wut, EH_, E_,  1.0f);

    // Q,K,V projections (WMMA)
    k_qkv<<<dim3(BT_ / 16, EH_ / 64, 3), dim3(32, 4), 0, stream>>>(
        Xbf, Wqt, Wkt, Wvt, Qbf, Kbf, Vtb);

    // fused masked-softmax attention (WMMA scores + WMMA attn@V)
    k_attn<<<dim3(T_ / 16, H_, B_), 256, 0, stream>>>(Qbf, Kbf, Vtb, mask, Obf);

    // output projection + bias (WMMA)
    k_final<<<dim3(BT_ / 16, E_ / 64), dim3(32, 4), 0, stream>>>(Obf, Wut, bu, out);
}
